// HyenaOperatorMR_62895501083285
// MI455X (gfx1250) — matrix operand order repaired and checked by reading, not compiled, not run on hardware
//
#include <hip/hip_runtime.h>
#include <hip/hip_bf16.h>
#include <stdint.h>

// ---------------------------------------------------------------------------
// Hyena-like operator on MI455X (gfx1250, wave32).
//   GEMM1 (16384x2304x768) and GEMM2 (16384x768x768):
//     v_wmma_f32_16x16x32_bf16 (fp32 accum) fed from double-buffered LDS that
//     is filled by the Tensor Data Mover (tensor_load_to_lds, TENSORcnt).
//   conv/elementwise stages stay fp32 (bandwidth-bound, ~600MB @ 23.3 TB/s).
// ---------------------------------------------------------------------------

typedef __attribute__((ext_vector_type(16))) __bf16 v16bf;
typedef __attribute__((ext_vector_type(8)))  float  v8f;
typedef __attribute__((ext_vector_type(4)))  unsigned int v4u;
typedef __attribute__((ext_vector_type(8)))  int    v8i;
typedef __attribute__((ext_vector_type(4)))  int    v4i;

union FragBF {       // 32 bytes = 16 bf16 = one WMMA A/B operand (8 VGPRs)
  uint4 u[2];
  v16bf bf;
};

#define BM 128
#define BN 128
#define BKK 32
#define FLEN 127

#if __has_builtin(__builtin_amdgcn_tensor_load_to_lds) && \
    __has_builtin(__builtin_amdgcn_s_wait_tensorcnt)
#define USE_TDM 1
#else
#define USE_TDM 0
#endif

#if USE_TDM
// ---------------------------------------------------------------------------
// Issue one TDM 2D tile load: tile_y rows x tile_x elements (2B each) from
// global (row stride = stride_elems) into LDS at lds_off, row-major packed.
// D# layout per CDNA5 ISA ch.8 (group0 128b, group1 256b; groups 2/3 unused
// for 2D tensors -> zero). This toolchain exposes the 6-arg builtin:
//   (v4u group0, v8i group1, v4i group2, v4i group3, v8i extra, i32 cpol)
// ---------------------------------------------------------------------------
__device__ __attribute__((always_inline)) void tdm_load_tile_2d(
    uint32_t lds_off, uint64_t gaddr, uint32_t tile_x, uint32_t tile_y,
    uint32_t stride_elems) {
  const uint32_t TD = 1u << 20;   // tensor dims (OOB bound; tiles never OOB)
  v4u g0;
  g0[0] = 1u;                                            // count=1, user D#
  g0[1] = lds_off;                                       // lds_addr [63:32]
  g0[2] = (uint32_t)gaddr;                               // global_addr lo
  g0[3] = ((uint32_t)(gaddr >> 32) & 0x01ffffffu)        // global_addr[56:32]
          | 0x80000000u;                                 // type=2 ("image")
  v8i g1;
  g1[0] = (int)0x00010000u;                              // data_size=1 (2B)
  g1[1] = (int)((TD & 0xffffu) << 16);                   // tensor_dim0 lo16
  g1[2] = (int)(((TD >> 16) & 0xffffu) |                 // tensor_dim0 hi16
                ((TD & 0xffffu) << 16));                 // tensor_dim1 lo16
  g1[3] = (int)(((TD >> 16) & 0xffffu) |                 // tensor_dim1 hi16
                (tile_x << 16));                         // tile_dim0
  g1[4] = (int)(tile_y & 0xffffu);                       // tile_dim1 (dim2=0)
  g1[5] = (int)stride_elems;                             // dim0 stride lo32
  g1[6] = 0;                                             // stride hi16, s1 lo
  g1[7] = 0;
  const v4i gz4 = {0, 0, 0, 0};
  const v8i gz8 = {0, 0, 0, 0, 0, 0, 0, 0};
  __builtin_amdgcn_tensor_load_to_lds(g0, g1, gz4, gz4, gz8, 0);
}
#endif

// ---------------------------------------------------------------------------
// fp32 -> bf16 grid-stride conversion
// ---------------------------------------------------------------------------
__global__ void cvt_f32_bf16(const float* __restrict__ in,
                             __bf16* __restrict__ out, long n) {
  long i = (long)blockIdx.x * blockDim.x + threadIdx.x;
  const long stride = (long)gridDim.x * blockDim.x;
  for (; i < n; i += stride) out[i] = (__bf16)in[i];
}

// ---------------------------------------------------------------------------
// C[M,N] (f32) = A[M,K](bf16, row-major) * B[N,K](bf16, row-major, i.e. W)^T
//             + bias[N]
// Block: 256 thr = 8 waves (2x4). Block tile 128x128, K-step 32, LDS double-
// buffered, tiles DMA'd by the TDM from wave 0. 8 WMMAs / wave / K-step.
// ---------------------------------------------------------------------------
__global__ __launch_bounds__(256) void gemm_bf16_wmma(
    const __bf16* __restrict__ A, const __bf16* __restrict__ Bw,
    const float* __restrict__ bias, float* __restrict__ C,
    int M, int N, int K) {
  __shared__ __align__(16) __bf16 lA[2][BM * BKK];   // [m][k]
  __shared__ __align__(16) __bf16 lB[2][BN * BKK];   // [n][k]

  const int tid  = threadIdx.x;
  const int lane = tid & 31;
  const int wave = tid >> 5;
  const int wm   = wave >> 2;          // 0..1  (M direction)
  const int wn   = wave & 3;           // 0..3  (N direction)
  const int half = lane >> 4;          // lane group 0 / 1
  const int l15  = lane & 15;
  const long m0  = (long)blockIdx.y * BM;
  const long n0  = (long)blockIdx.x * BN;

  const uint64_t gaA = (uint64_t)(uintptr_t)(A  + m0 * (long)K);
  const uint64_t gaB = (uint64_t)(uintptr_t)(Bw + n0 * (long)K);

  const v8f vzero = {};
  v8f acc[4][2];
#pragma unroll
  for (int i = 0; i < 4; ++i)
#pragma unroll
    for (int j = 0; j < 2; ++j) acc[i][j] = vzero;

#if USE_TDM
  if (wave == 0) {           // prologue: DMA first tiles into buffer 0
    tdm_load_tile_2d((uint32_t)(uintptr_t)&lA[0][0], gaA, BKK, BM, K);
    tdm_load_tile_2d((uint32_t)(uintptr_t)&lB[0][0], gaB, BKK, BN, K);
  }
#endif

  int nb = 0;
  for (int k0 = 0; k0 < K; k0 += BKK, nb ^= 1) {
#if USE_TDM
    if (wave == 0) __builtin_amdgcn_s_wait_tensorcnt(0);
    __syncthreads();         // buffer nb visible; buffer nb^1 compute done
    if (wave == 0 && (k0 + BKK) < K) {
      tdm_load_tile_2d((uint32_t)(uintptr_t)&lA[nb ^ 1][0],
                       gaA + (uint64_t)(k0 + BKK) * 2, BKK, BM, K);
      tdm_load_tile_2d((uint32_t)(uintptr_t)&lB[nb ^ 1][0],
                       gaB + (uint64_t)(k0 + BKK) * 2, BKK, BN, K);
    }
#else
    {  // fallback staging: 2 threads per row, 32B each, batched loads
      const int lr = tid >> 1, lh = tid & 1;
      const uint4* ga = (const uint4*)((const __bf16*)(uintptr_t)gaA +
                                       (long)lr * K + k0 + lh * 16);
      const uint4* gb = (const uint4*)((const __bf16*)(uintptr_t)gaB +
                                       (long)lr * K + k0 + lh * 16);
      uint4 a0 = ga[0], a1 = ga[1], b0 = gb[0], b1 = gb[1];
      __syncthreads();
      uint4* sa = (uint4*)&lA[nb][lr * BKK + lh * 16];
      uint4* sb = (uint4*)&lB[nb][lr * BKK + lh * 16];
      sa[0] = a0; sa[1] = a1; sb[0] = b0; sb[1] = b1;
      __syncthreads();
    }
#endif
    const __bf16* bufA = lA[nb];
    const __bf16* bufB = lB[nb];

    // B fragments: lane holds column N=l15; K = half*16 .. half*16+15
    FragBF bf[2];
#pragma unroll
    for (int j = 0; j < 2; ++j) {
      const int n = wn * 32 + j * 16 + l15;
      const uint4* p = (const uint4*)&bufB[n * BKK + half * 16];
      bf[j].u[0] = p[0];
      bf[j].u[1] = p[1];
    }
    // A fragments: lane holds row M=l15; K = half*16 .. half*16+15
#pragma unroll
    for (int i = 0; i < 4; ++i) {
      FragBF af;
      const int r = wm * 64 + i * 16 + l15;
      af.u[0] = *(const uint4*)&bufA[r * BKK + half * 16];
      af.u[1] = *(const uint4*)&bufA[r * BKK + half * 16 + 8];
#pragma unroll
      for (int j = 0; j < 2; ++j) {
        acc[i][j] = __builtin_amdgcn_wmma_f32_16x16x32_bf16(
            false, af.bf, false, bf[j].bf, (short)0, acc[i][j], false, false);
      }
    }
#if USE_TDM
    __syncthreads();  // compute on nb done before TDM (next iter) reuses it
#endif
  }

  // epilogue: D VGPR v holds row M = half*8 + v, col N = l15 (coalesced)
#pragma unroll
  for (int i = 0; i < 4; ++i) {
#pragma unroll
    for (int j = 0; j < 2; ++j) {
      const long n  = n0 + wn * 32 + j * 16 + l15;
      const float bv = bias ? bias[n] : 0.0f;
      const long mb = m0 + wm * 64 + i * 16 + half * 8;
#pragma unroll
      for (int v = 0; v < 8; ++v) {
        C[(mb + v) * (long)N + n] = acc[i][j][v] + bv;
      }
    }
  }
}

// ---------------------------------------------------------------------------
// Short depthwise conv (K=3, pad=1) over L on H[B,L,3D], then:
//   C0_cm[b][d][l] = conv(chunk0), V_cm[b][d][l] = conv(chunk1)*conv(chunk2)
// ---------------------------------------------------------------------------
__global__ __launch_bounds__(256) void shortconv_kernel(
    const float* __restrict__ H, const float* __restrict__ sw,
    const float* __restrict__ sb, float* __restrict__ C0,
    float* __restrict__ V) {
  const int d = blockIdx.x * 256 + threadIdx.x;   // 0..767
  const int l = blockIdx.y;                       // 0..2047
  const int b = blockIdx.z;                       // 0..7
  const long row  = (long)(b * 2048 + l) * 2304;
  const bool hasM = (l > 0), hasP = (l < 2047);
  float r[3];
#pragma unroll
  for (int q = 0; q < 3; ++q) {
    const int c = q * 768 + d;
    float a = sb[c];
    const float w0 = sw[c * 3 + 0];
    const float w1 = sw[c * 3 + 1];
    const float w2 = sw[c * 3 + 2];
    if (hasM) a += H[row - 2304 + c] * w0;
    a += H[row + c] * w1;
    if (hasP) a += H[row + 2304 + c] * w2;
    r[q] = a;
  }
  const long cm = ((long)(b * 768 + d)) * 2048 + l;
  C0[cm] = r[0];
  V[cm]  = r[1] * r[2];
}

// ---------------------------------------------------------------------------
// Long depthwise conv (FL=127, pad=63) per (b,d), filter = long_f * analytic
// decay window; fused with *C0, SiLU, bf16 pack for GEMM2.
// ---------------------------------------------------------------------------
__global__ __launch_bounds__(256) void longconv_silu_kernel(
    const float* __restrict__ V, const float* __restrict__ C0,
    const float* __restrict__ lf, __bf16* __restrict__ S) {
  __shared__ float sv[2048 + 2 * 63];
  __shared__ float fw[FLEN];
  const int d = blockIdx.x;   // 0..767
  const int b = blockIdx.y;   // 0..7
  const int t = threadIdx.x;
  const long base = ((long)(b * 768 + d)) * 2048;

  // rate_d = 10^(-1 + 3d/767); win_j = exp(-rate*|j-63|/63) + 0.05
  const float rate =
      __expf(2.302585093f * (-1.0f + 3.0f * (float)d * (1.0f / 767.0f)));
  for (int j = t; j < FLEN; j += 256) {
    const float tt = fabsf((float)(j - 63)) * (1.0f / 63.0f);
    fw[j] = lf[d * FLEN + j] * (__expf(-rate * tt) + 0.05f);
  }
  for (int l = t; l < 2048; l += 256) sv[63 + l] = V[base + l];
  for (int j = t; j < 63; j += 256) { sv[j] = 0.0f; sv[63 + 2048 + j] = 0.0f; }
  __syncthreads();

  for (int l = t; l < 2048; l += 256) {
    float acc = 0.0f;
#pragma unroll 4
    for (int j = 0; j < FLEN; ++j) acc += sv[l + j] * fw[j];
    const float g = acc * C0[base + l];
    const float s = g / (1.0f + __expf(-g));          // silu
    S[((long)(b * 2048 + l)) * 768 + d] = (__bf16)s;  // row-major for GEMM2
  }
}

// ---------------------------------------------------------------------------
// Workspace layout (bytes); Sb aliases Xb (dead after GEMM1). Total ~269 MB.
// ---------------------------------------------------------------------------
#define XB_OFF   0L                                 // 16384*768  bf16 = 25165824
#define WIB_OFF  25165824L                          // 2304*768   bf16 =  3538944
#define WOB_OFF  28704768L                          // 768*768    bf16 =  1179648
#define H_OFF    29884416L                          // 16384*2304 f32  = 150994944
#define C0_OFF   180879360L                         // 8*768*2048 f32  = 50331648
#define V_OFF    231211008L                         // 8*768*2048 f32  = 50331648
#define SB_OFF   XB_OFF                             // 16384*768  bf16 (alias)

extern "C" void kernel_launch(void* const* d_in, const int* in_sizes, int n_in,
                              void* d_out, int out_size, void* d_ws, size_t ws_size,
                              hipStream_t stream) {
  const float* x       = (const float*)d_in[0];
  const float* W_in    = (const float*)d_in[1];
  const float* b_in    = (const float*)d_in[2];
  const float* short_w = (const float*)d_in[3];
  const float* short_b = (const float*)d_in[4];
  const float* long_f  = (const float*)d_in[5];
  const float* W_out   = (const float*)d_in[6];
  const float* b_out   = (const float*)d_in[7];
  float* out = (float*)d_out;

  char* ws = (char*)d_ws;
  __bf16* Xb  = (__bf16*)(ws + XB_OFF);
  __bf16* Wib = (__bf16*)(ws + WIB_OFF);
  __bf16* Wob = (__bf16*)(ws + WOB_OFF);
  float*  H   = (float*)(ws + H_OFF);
  float*  C0  = (float*)(ws + C0_OFF);
  float*  V   = (float*)(ws + V_OFF);
  __bf16* Sb  = (__bf16*)(ws + SB_OFF);

  // 1. bf16 casts for the matrix pipes
  cvt_f32_bf16<<<4096, 256, 0, stream>>>(x,     Xb,  16384L * 768);
  cvt_f32_bf16<<<2048, 256, 0, stream>>>(W_in,  Wib, 2304L  * 768);
  cvt_f32_bf16<<<1024, 256, 0, stream>>>(W_out, Wob, 768L   * 768);

  // 2. GEMM1: H[16384,2304] = Xb @ W_in^T + b_in
  gemm_bf16_wmma<<<dim3(2304 / BN, 16384 / BM), 256, 0, stream>>>(
      Xb, Wib, b_in, H, 16384, 2304, 768);

  // 3. short depthwise conv + chunk products -> C0, V (channel-major)
  shortconv_kernel<<<dim3(3, 2048, 8), 256, 0, stream>>>(
      H, short_w, short_b, C0, V);

  // 4. long depthwise conv * decay window, fused *C0, SiLU, bf16 pack
  longconv_silu_kernel<<<dim3(768, 8), 256, 0, stream>>>(V, C0, long_f, Sb);

  // 5. GEMM2: out[16384,768] = Sb @ W_out^T + b_out
  gemm_bf16_wmma<<<dim3(768 / BN, 16384 / BM), 256, 0, stream>>>(
      Sb, Wob, b_out, out, 16384, 768, 768);
}
